// E_GCL_multi_mes_88476326297728
// MI455X (gfx1250) — compile-verified, matrix operand order checked by reading, not run on hardware
//
#include <hip/hip_runtime.h>
#include <hip/hip_bf16.h>

#define MAX_ATOM 54
#define B_MOL 1024
#define N_NODE (B_MOL * MAX_ATOM)
#define DEG 16
#define N_EDGE (N_NODE * DEG)
#define HF 64
#define MT 16

typedef __bf16 bf16_t;
typedef bf16_t v16bf __attribute__((ext_vector_type(16)));
typedef float v8f __attribute__((ext_vector_type(8)));
typedef unsigned short v16u __attribute__((ext_vector_type(16)));

struct U32x8 { uint4 lo, hi; };

// Packed-weight fragment offsets inside d_ws (units of 512 ushorts = one frag).
// edge_w1 is repacked to K=224 so its rows line up with the in-place staging
// buffer layout [edge_feat(0..63) | radial(64) | dead(65..128) | ea(129..192) | 0].
enum {
  FRAG_MES1 = 0,    // K 193->224 : 7*4 = 28
  FRAG_MES2 = 28,   // K 64 : 8
  FRAG_CW1  = 36,   // K 64 : 8
  FRAG_EW1  = 44,   // K 224 (remapped 129) : 28
  FRAG_EW2  = 72,   // K 64 : 8
  FRAG_NW1  = 80,   // K 128 : 16
  FRAG_NW2  = 96,   // K 64 : 8
  FRAG_TOTAL = 104
};

__device__ __forceinline__ unsigned short f2bf(float f) {
  unsigned int u = __float_as_uint(f);
  u += 0x7fffu + ((u >> 16) & 1u);   // round-to-nearest-even
  return (unsigned short)(u >> 16);
}

// Fast silu: v_exp + v_rcp instead of IEEE-div expansion.
__device__ __forceinline__ float siluf(float x) {
  return x * __builtin_amdgcn_rcpf(1.0f + __expf(-x));
}

// B fragment: prepacked in ws so each lane reads 32 contiguous bytes.
__device__ __forceinline__ v16bf load_b_frag(const unsigned short* __restrict__ wsb,
                                             int frag, int lane) {
  const uint4* p = (const uint4*)(wsb + (size_t)frag * 512 + (size_t)lane * 16);
  U32x8 u; u.lo = p[0]; u.hi = p[1];
  return __builtin_bit_cast(v16bf, u);
}

// A fragment from a bf16 LDS tile (row-major, 16 rows x stride cols).
// lane<16 -> M=lane, K in {kt*32+0..7, kt*32+16..23};
// lane>=16 -> M=lane-16, K in {kt*32+8..15, kt*32+24..31}.
__device__ __forceinline__ v16bf load_a_lds(const unsigned short* buf, int stride,
                                            int kt, int lane) {
  int m = lane & 15;
  int kb = (lane & 16) ? 8 : 0;
  const unsigned short* p = buf + m * stride + kt * 32;
  U32x8 u;
  u.lo = *(const uint4*)(p + kb);
  u.hi = *(const uint4*)(p + 16 + kb);
  return __builtin_bit_cast(v16bf, u);
}

template <int KT>
__device__ __forceinline__ void gemm_lds(v8f acc[4], const unsigned short* abuf, int astride,
                                         const unsigned short* __restrict__ wsb,
                                         int frag_base, int lane) {
#pragma unroll
  for (int kt = 0; kt < KT; ++kt) {
    v16bf a = load_a_lds(abuf, astride, kt, lane);
#pragma unroll
    for (int nt = 0; nt < 4; ++nt) {
      v16bf b = load_b_frag(wsb, frag_base + kt * 4 + nt, lane);
      acc[nt] = __builtin_amdgcn_wmma_f32_16x16x32_bf16(
          false, a, false, b, (short)0, acc[nt], false, false);
    }
  }
}

// Bias splat from 4 pre-loaded per-lane scalars.
__device__ __forceinline__ void init_bias_r(v8f acc[4], const float bs[4]) {
#pragma unroll
  for (int nt = 0; nt < 4; ++nt)
#pragma unroll
    for (int i = 0; i < 8; ++i) acc[nt][i] = bs[nt];
}

__device__ __forceinline__ void silu_acc(v8f acc[4]) {
#pragma unroll
  for (int nt = 0; nt < 4; ++nt)
#pragma unroll
    for (int i = 0; i < 8; ++i) acc[nt][i] = siluf(acc[nt][i]);
}

// Store C frags to a 16-row bf16 LDS tile with arbitrary row stride.
__device__ __forceinline__ void store_c_bf16(unsigned short* buf, int stride,
                                             const v8f acc[4], int lane) {
  int n = lane & 15;
  int mh = (lane & 16) ? 8 : 0;
#pragma unroll
  for (int nt = 0; nt < 4; ++nt)
#pragma unroll
    for (int v = 0; v < 8; ++v)
      buf[(v + mh) * stride + nt * 16 + n] = f2bf(acc[nt][v]);
}

// ---------------------------------------------------------------------------
// Prep kernel: convert weight matrices to bf16 WMMA-B fragment layout.
// ---------------------------------------------------------------------------
__global__ void pack_weights(const float* __restrict__ mes_w1, const float* __restrict__ mes_w2,
                             const float* __restrict__ coord_w1,
                             const float* __restrict__ edge_w1, const float* __restrict__ edge_w2,
                             const float* __restrict__ node_w1, const float* __restrict__ node_w2,
                             unsigned short* __restrict__ ws) {
  int f = blockIdx.x;
  int lane = threadIdx.x;
  const float* w; int Kreal; int local; bool remap = false;
  if (f < FRAG_MES2)      { w = mes_w1;   Kreal = 193; local = f - FRAG_MES1; }
  else if (f < FRAG_CW1)  { w = mes_w2;   Kreal = 64;  local = f - FRAG_MES2; }
  else if (f < FRAG_EW1)  { w = coord_w1; Kreal = 64;  local = f - FRAG_CW1; }
  else if (f < FRAG_EW2)  { w = edge_w1;  Kreal = 129; local = f - FRAG_EW1; remap = true; }
  else if (f < FRAG_NW1)  { w = edge_w2;  Kreal = 64;  local = f - FRAG_EW2; }
  else if (f < FRAG_NW2)  { w = node_w1;  Kreal = 128; local = f - FRAG_NW1; }
  else                    { w = node_w2;  Kreal = 64;  local = f - FRAG_NW2; }
  int kt = local >> 2, nt = local & 3;
  int n = nt * 16 + (lane & 15);
  int kb = (lane & 16) ? 8 : 0;
  unsigned short* out = ws + (size_t)f * 512 + (size_t)lane * 16;
#pragma unroll
  for (int i = 0; i < 16; ++i) {
    int k = kt * 32 + ((i < 8) ? (kb + i) : (16 + kb + (i - 8)));
    float v = 0.0f;
    if (remap) {
      // edge_w1 row k' of staged edge_in: [feat 0..63 | radial 64 | dead | ea 129..192]
      if (k < 65) v = w[(size_t)k * HF + n];
      else if (k >= 129 && k < 193) v = w[(size_t)(65 + (k - 129)) * HF + n];
    } else if (k < Kreal) {
      v = w[(size_t)k * HF + n];
    }
    out[i] = f2bf(v);
  }
}

// ---------------------------------------------------------------------------
// Main kernel: one block per molecule, 4 wave32s. Each wave owns atom-tiles
// (an atom's 16 contiguous edges == one WMMA M=16 tile).
// __launch_bounds__(128, 2): cap VGPRs so >=2 waves/SIMD stay resident; B
// fragments re-load from WGP$/L2 instead of being pinned in 800+ VGPRs.
// ---------------------------------------------------------------------------
__global__ __launch_bounds__(128, 2) void egnn_mol_kernel(
    const float* __restrict__ h, const float* __restrict__ coord,
    const float* __restrict__ edge_attr, const float* __restrict__ jt_mess,
    const float* __restrict__ node_embed, const float* __restrict__ node_mask,
    const float* __restrict__ edge_mask,
    const float* __restrict__ mes_b1, const float* __restrict__ mes_b2,
    const float* __restrict__ edge_b1, const float* __restrict__ edge_b2,
    const float* __restrict__ node_b1, const float* __restrict__ node_b2,
    const float* __restrict__ coord_b1, const float* __restrict__ coord_w2,
    const int* __restrict__ edge_index, const int* __restrict__ mess_holder,
    const int* __restrict__ node_holder, const int* __restrict__ jt_idx,
    const unsigned short* __restrict__ wfrag,
    float* __restrict__ out) {
  __shared__ __align__(16) unsigned short sh_hb[MAX_ATOM * HF];  // h + node_collect, bf16
  __shared__ float sh_agg[MAX_ATOM * HF];                        // segment_sum(edge_feat, col)
  __shared__ __align__(16) unsigned short sA[4][16 * 224];       // per-wave staging
  __shared__ __align__(16) unsigned short sT[4][16 * HF];        // per-wave activations
  __shared__ int   sh_ca[4][16];
  __shared__ int   sh_jm[4][16];
  __shared__ float sh_em[4][16];
  __shared__ float sh_rad[4][16];
  __shared__ float sh_cdn[4][48];
  __shared__ float sh_gate[4][16];
  __shared__ float sh_tr[4][4];

  const int cand = blockIdx.x;
  const int tid = threadIdx.x;
  const int w = tid >> 5;
  const int lane = tid & 31;
  const int jt = jt_idx[cand];
  const size_t jtbase = (size_t)jt * MT * MT * HF;

  const size_t out_coord = (size_t)N_NODE * HF;
  const size_t out_edge = out_coord + (size_t)N_NODE * 3;

  // Phase 0: h_new = h + node_collect (bf16 in LDS); zero agg.
  for (int idx = tid; idx < MAX_ATOM * HF; idx += 128) {
    int a = idx >> 6, c = idx & 63;
    int g = cand * MAX_ATOM + a;
    int jn = node_holder[g];
    int jn0 = (jn - 1 > 0) ? (jn - 1) : 0;
    float v = h[(size_t)g * HF + c] + node_embed[((size_t)jt * MT + jn0) * HF + c];
    sh_hb[idx] = f2bf(v);
    sh_agg[idx] = 0.0f;
  }
  __syncthreads();

  unsigned short* sAw = sA[w];
  unsigned short* sTw = sT[w];

  // Hoist per-lane bias scalars for the edge-phase layers.
  const int n0 = lane & 15;
  float bs_mes1[4], bs_mes2[4], bs_cw1[4], bs_ew1[4], bs_ew2[4], cw2r[4];
#pragma unroll
  for (int nt = 0; nt < 4; ++nt) {
    bs_mes1[nt] = mes_b1[nt * 16 + n0];
    bs_mes2[nt] = mes_b2[nt * 16 + n0];
    bs_cw1[nt]  = coord_b1[nt * 16 + n0];
    bs_ew1[nt]  = edge_b1[nt * 16 + n0];
    bs_ew2[nt]  = edge_b2[nt * 16 + n0];
    cw2r[nt]    = coord_w2[nt * 16 + n0];
  }

  // Prefetch this wave's first edge_attr tile (32 lanes x 128B = 4KB tile).
  {
    const char* p = (const char*)(edge_attr + (size_t)(cand * MAX_ATOM + w) * DEG * HF);
    __builtin_prefetch(p + lane * 128, 0, 1);
  }

  // -------------------- Edge phase: one atom-tile per wave iteration ------
  for (int a = w; a < MAX_ATOM; a += 4) {
    const int g_atom = cand * MAX_ATOM + a;

    // Prefetch next tile's edge_attr while we compute this one.
    if (a + 4 < MAX_ATOM) {
      const char* p = (const char*)(edge_attr + (size_t)(g_atom + 4) * DEG * HF);
      __builtin_prefetch(p + lane * 128, 0, 1);
    }

    if (lane < 16) {
      int e = g_atom * DEG + lane;
      int colg = edge_index[N_EDGE + e];
      int ca = colg - cand * MAX_ATOM;  // col within molecule
      int mh = ((cand * MAX_ATOM + a) * MAX_ATOM + ca) * 2;
      int jr = mess_holder[mh], jc = mess_holder[mh + 1];
      int jm = (jr != 0 && jc != 0) ? (((jr - 1) * MT + (jc - 1)) * HF) : -1;
      float dx = coord[(size_t)g_atom * 3 + 0] - coord[(size_t)colg * 3 + 0];
      float dy = coord[(size_t)g_atom * 3 + 1] - coord[(size_t)colg * 3 + 1];
      float dz = coord[(size_t)g_atom * 3 + 2] - coord[(size_t)colg * 3 + 2];
      float rad = dx * dx + dy * dy + dz * dz;
      float inv = 1.0f / (sqrtf(rad + 1e-8f) + 1.0f);
      sh_ca[w][lane] = ca;
      sh_jm[w][lane] = jm;
      sh_em[w][lane] = edge_mask[e];
      sh_rad[w][lane] = rad;
      sh_cdn[w][lane * 3 + 0] = dx * inv;
      sh_cdn[w][lane * 3 + 1] = dy * inv;
      sh_cdn[w][lane * 3 + 2] = dz * inv;
    }

    // Stage mes_in = [h_row | h_col | radial | edge_attr+mess | pad] (16 x 224 bf16).
#pragma unroll 1
    for (int el = 0; el < 16; ++el) {
      int ca = sh_ca[w][el];
      int jm = sh_jm[w][el];
      unsigned short radb = f2bf(sh_rad[w][el]);
      size_t ebase = (size_t)(g_atom * DEG + el) * HF;
      unsigned short* rowp = sAw + el * 224;
#pragma unroll
      for (int c = lane; c < 224; c += 32) {
        unsigned short val = 0;
        if (c < 64) val = sh_hb[a * HF + c];
        else if (c < 128) val = sh_hb[ca * HF + (c - 64)];
        else if (c == 128) val = radb;
        else if (c < 193) {
          int j = c - 129;
          float ea = edge_attr[ebase + j];
          if (jm >= 0) ea += jt_mess[jtbase + jm + j];
          val = f2bf(ea);
        }
        rowp[c] = val;
      }
    }

    v8f acc[4];
    // mes MLP layer 1: 16x224 @ 224x64
    init_bias_r(acc, bs_mes1);
    gemm_lds<7>(acc, sAw, 224, wfrag, FRAG_MES1, lane);
    silu_acc(acc);
    store_c_bf16(sTw, HF, acc, lane);

    // mes MLP layer 2 (+ edge_mask): edge_feat
    init_bias_r(acc, bs_mes2);
    gemm_lds<2>(acc, sTw, HF, wfrag, FRAG_MES2, lane);
    {
      int mh = (lane & 16) ? 8 : 0;
      float em[8]; int cam[8];
#pragma unroll
      for (int v = 0; v < 8; ++v) { em[v] = sh_em[w][v + mh]; cam[v] = sh_ca[w][v + mh]; }
#pragma unroll
      for (int nt = 0; nt < 4; ++nt)
#pragma unroll
        for (int v = 0; v < 8; ++v) {
          float x = siluf(acc[nt][v]) * em[v];
          acc[nt][v] = x;
          atomicAdd(&sh_agg[cam[v] * HF + nt * 16 + n0], x);  // agg over col
        }
    }
    // Overwrite dead mes_in cols 0..64 of sAw with edge_feat | radial (in-place edge_in).
    store_c_bf16(sAw, 224, acc, lane);
    if (lane < 16) sAw[lane * 224 + 64] = f2bf(sh_rad[w][lane]);

    // gate = silu(edge_feat @ coord_w1 + b) @ coord_w2 (A = sAw cols 0..63)
    {
      v8f g4[4];
      init_bias_r(g4, bs_cw1);
      gemm_lds<2>(g4, sAw, 224, wfrag, FRAG_CW1, lane);
      int mh = (lane & 16) ? 8 : 0;
      float p[8];
#pragma unroll
      for (int v = 0; v < 8; ++v) p[v] = 0.0f;
#pragma unroll
      for (int nt = 0; nt < 4; ++nt) {
#pragma unroll
        for (int v = 0; v < 8; ++v) p[v] += siluf(g4[nt][v]) * cw2r[nt];
      }
      if (lane < 16) sh_gate[w][lane] = 0.0f;
#pragma unroll
      for (int v = 0; v < 8; ++v) atomicAdd(&sh_gate[w][v + mh], p[v]);
    }
    // coord: all 16 edges of this tile share the same row -> tile reduction
    if (lane < 4) sh_tr[w][lane] = 0.0f;
    if (lane < 16) {
      float tf = sh_gate[w][lane] * sh_em[w][lane];
      atomicAdd(&sh_tr[w][0], sh_cdn[w][lane * 3 + 0] * tf);
      atomicAdd(&sh_tr[w][1], sh_cdn[w][lane * 3 + 1] * tf);
      atomicAdd(&sh_tr[w][2], sh_cdn[w][lane * 3 + 2] * tf);
    }
    if (lane == 0) {
      float nm = node_mask[g_atom];
#pragma unroll
      for (int k = 0; k < 3; ++k)
        out[out_coord + (size_t)g_atom * 3 + k] =
            (coord[(size_t)g_atom * 3 + k] + sh_tr[w][k]) * nm;
    }

    // edge-out MLP layer 1 over staged edge_in (K=224, zero weights on dead cols)
    init_bias_r(acc, bs_ew1);
    gemm_lds<7>(acc, sAw, 224, wfrag, FRAG_EW1, lane);
    silu_acc(acc);
    store_c_bf16(sTw, HF, acc, lane);

    // edge-out MLP layer 2 (+ edge_mask) -> global edge_attr output
    init_bias_r(acc, bs_ew2);
    gemm_lds<2>(acc, sTw, HF, wfrag, FRAG_EW2, lane);
    {
      int mh = (lane & 16) ? 8 : 0;
#pragma unroll
      for (int nt = 0; nt < 4; ++nt)
#pragma unroll
        for (int v = 0; v < 8; ++v) {
          int m = v + mh;
          int e = g_atom * DEG + m;
          out[out_edge + (size_t)e * HF + nt * 16 + n0] = acc[nt][v] * sh_em[w][m];
        }
    }
  }

  __syncthreads();  // agg complete

  // -------------------- Node phase: wave w -> rows [w*16, w*16+16) --------
  {
    // Stage node_in = [h_new | agg] (16 x 128 bf16) into this wave's sA region.
#pragma unroll 1
    for (int m = 0; m < 16; ++m) {
      int r = w * 16 + m;
      unsigned short* rowp = sAw + m * 128;
#pragma unroll
      for (int c = lane; c < 128; c += 32) {
        unsigned short val = 0;
        if (r < MAX_ATOM)
          val = (c < HF) ? sh_hb[r * HF + c] : f2bf(sh_agg[r * HF + (c - HF)]);
        rowp[c] = val;
      }
    }

    float bs_nw1[4], bs_nw2[4];
#pragma unroll
    for (int nt = 0; nt < 4; ++nt) {
      bs_nw1[nt] = node_b1[nt * 16 + n0];
      bs_nw2[nt] = node_b2[nt * 16 + n0];
    }

    v8f acc[4];
    init_bias_r(acc, bs_nw1);
    gemm_lds<4>(acc, sAw, 128, wfrag, FRAG_NW1, lane);
    silu_acc(acc);
    store_c_bf16(sTw, HF, acc, lane);

    init_bias_r(acc, bs_nw2);
    gemm_lds<2>(acc, sTw, HF, wfrag, FRAG_NW2, lane);
    {
      int mh = (lane & 16) ? 8 : 0;
#pragma unroll
      for (int nt = 0; nt < 4; ++nt)
#pragma unroll
        for (int v = 0; v < 8; ++v) {
          int r = w * 16 + v + mh;
          if (r < MAX_ATOM) {
            int g = cand * MAX_ATOM + r;
            int c = nt * 16 + n0;
            int jn = node_holder[g];
            int jn0 = (jn - 1 > 0) ? (jn - 1) : 0;
            float base = h[(size_t)g * HF + c] +
                         node_embed[((size_t)jt * MT + jn0) * HF + c];
            out[(size_t)g * HF + c] = (base + acc[nt][v]) * node_mask[g];
          }
        }
    }
  }
}

extern "C" void kernel_launch(void* const* d_in, const int* in_sizes, int n_in,
                              void* d_out, int out_size, void* d_ws, size_t ws_size,
                              hipStream_t stream) {
  (void)in_sizes; (void)n_in; (void)out_size; (void)ws_size;
  const float* h          = (const float*)d_in[0];
  const float* coord      = (const float*)d_in[1];
  const float* edge_attr  = (const float*)d_in[2];
  const float* jt_mess    = (const float*)d_in[3];
  const float* node_embed = (const float*)d_in[4];
  const float* node_mask  = (const float*)d_in[5];
  const float* edge_mask  = (const float*)d_in[6];
  const float* mes_w1     = (const float*)d_in[7];
  const float* mes_b1     = (const float*)d_in[8];
  const float* mes_w2     = (const float*)d_in[9];
  const float* mes_b2     = (const float*)d_in[10];
  const float* edge_w1    = (const float*)d_in[11];
  const float* edge_b1    = (const float*)d_in[12];
  const float* edge_w2    = (const float*)d_in[13];
  const float* edge_b2    = (const float*)d_in[14];
  const float* node_w1    = (const float*)d_in[15];
  const float* node_b1    = (const float*)d_in[16];
  const float* node_w2    = (const float*)d_in[17];
  const float* node_b2    = (const float*)d_in[18];
  const float* coord_w1   = (const float*)d_in[19];
  const float* coord_b1   = (const float*)d_in[20];
  const float* coord_w2   = (const float*)d_in[21];
  const int* edge_index   = (const int*)d_in[22];
  const int* mess_holder  = (const int*)d_in[23];
  const int* node_holder  = (const int*)d_in[24];
  const int* jt_idx       = (const int*)d_in[25];
  unsigned short* wfrag   = (unsigned short*)d_ws;

  pack_weights<<<FRAG_TOTAL, 32, 0, stream>>>(mes_w1, mes_w2, coord_w1, edge_w1, edge_w2,
                                              node_w1, node_w2, wfrag);
  egnn_mol_kernel<<<B_MOL, 128, 0, stream>>>(
      h, coord, edge_attr, jt_mess, node_embed, node_mask, edge_mask,
      mes_b1, mes_b2, edge_b1, edge_b2, node_b1, node_b2, coord_b1, coord_w2,
      edge_index, mess_holder, node_holder, jt_idx, wfrag, (float*)d_out);
}
